// NetworkODEModel_50921132261487
// MI455X (gfx1250) — matrix operand order repaired and checked by reading, not compiled, not run on hardware
//
#include <hip/hip_runtime.h>
#include <math.h>

// Problem constants (fixed by the reference)
#define B_  16
#define N_  512
#define HN  64
#define HC  16

typedef __attribute__((ext_vector_type(16))) _Float16 v16h;
typedef __attribute__((ext_vector_type(2)))  _Float16 v2h;
typedef __attribute__((ext_vector_type(8)))  float    v8f;

__device__ __forceinline__ float leaky(float v) { return fmaxf(v, 0.01f * v); }

// ---------------------------------------------------------------------------
// Kernel 1: node MLP (2 -> 64 -> 64 -> 1) per (b,i) thread + out0 = x[...,1].
// ---------------------------------------------------------------------------
__global__ void node_mlp_kernel(const float* __restrict__ x,
                                const float* __restrict__ Wn1, const float* __restrict__ bn1,
                                const float* __restrict__ Wn2, const float* __restrict__ bn2,
                                const float* __restrict__ Wn3, const float* __restrict__ bn3,
                                float* __restrict__ node_ws, float* __restrict__ out)
{
    int t = blockIdx.x * blockDim.x + threadIdx.x;      // 0 .. B*N-1
    float x0 = x[2 * t + 0];
    float x1 = x[2 * t + 1];

    float h[HN];
    #pragma unroll
    for (int u = 0; u < HN; ++u)
        h[u] = leaky(fmaf(x0, Wn1[u], fmaf(x1, Wn1[HN + u], bn1[u])));

    float o = bn3[0];
    for (int v = 0; v < HN; ++v) {
        float s = bn2[v];
        #pragma unroll
        for (int u = 0; u < HN; ++u)
            s = fmaf(h[u], Wn2[u * HN + v], s);
        o = fmaf(leaky(s), Wn3[v], o);
    }
    node_ws[t]     = o;      // node_out[b,i,0]
    out[2 * t + 0] = x1;     // out0 = x[..., 1]
}

// ---------------------------------------------------------------------------
// Kernel 2: fused pairwise coupling. One wave32 per (b, i).
//   Phase 0: sigmoid(A[i,:]) once per wave -> LDS (v_exp + v_rcp), rowsum
//            folded in for the bc3 term.
//   Loop:    j in 16-row tiles. h1 fragment built with packed-f16 FMAs
//            (v_pk_fma_f16 / v_pk_max_f16, no f32->f16 converts). bc2 is
//            folded into the WMMA C accumulator. Epilogue: leaky as
//            v_mul+v_max, sg from LDS broadcast, Wc3[n] hoisted out of
//            the loop entirely (lane-constant).
// ---------------------------------------------------------------------------
__global__ void __launch_bounds__(256)
coupling_kernel(const float* __restrict__ x,
                const float* __restrict__ Wc1, const float* __restrict__ bc1,
                const float* __restrict__ Wc2, const float* __restrict__ bc2,
                const float* __restrict__ Wc3, const float* __restrict__ bc3,
                const float* __restrict__ A_param,
                const float* __restrict__ node_ws,
                float* __restrict__ out)
{
    __shared__ float sgrow[8 * N_];                     // 8 waves * 512 * 4B = 16 KB

    const int lane  = threadIdx.x & 31;
    const int wslot = threadIdx.x >> 5;                 // wave within block
    const int wave  = (blockIdx.x * blockDim.x + threadIdx.x) >> 5;  // 0 .. B*N-1
    const int b     = wave >> 9;                        // wave / N
    const int i     = wave & (N_ - 1);                  // wave % N
    const int n     = lane & 15;                        // B/C/D column, A-frag row
    const int half  = lane >> 4;                        // K-chunk selector (A frag)
    const int c0    = half * 8;                         // this lane's channel base

    float* __restrict__ srow = &sgrow[wslot * N_];

    // ---- Phase 0: sigmoid(A[i,:]) row -> LDS; rowsum for bc3 term ------
    float accA = 0.f;
    #pragma unroll
    for (int t = 0; t < 16; ++t) {
        int   j  = lane + 32 * t;                       // coalesced over the wave
        float av = A_param[i * N_ + j] - (j == i ? 1e5f : 0.f);   // - eye/EPS
        float sg = __builtin_amdgcn_rcpf(1.f + __expf(-av));      // fast sigmoid
        srow[j] = sg;
        accA   += sg;
    }
    // Same wave writes & reads its own LDS segment; per-wave LDS ops are
    // in-order, no barrier needed.

    // ---- per-wave constants (packed f16 pairs for the fragment build) --
    const float xi0 = x[(b * N_ + i) * 2 + 0];
    const float xi1 = x[(b * N_ + i) * 2 + 1];

    v2h baseh[4], wj0h[4], wj1h[4];
    #pragma unroll
    for (int p = 0; p < 4; ++p) {
        int c = c0 + 2 * p;
        float b0 = fmaf(xi0, Wc1[2 * HC + c],     fmaf(xi1, Wc1[3 * HC + c],     bc1[c]));
        float b1 = fmaf(xi0, Wc1[2 * HC + c + 1], fmaf(xi1, Wc1[3 * HC + c + 1], bc1[c + 1]));
        baseh[p] = (v2h){(_Float16)b0, (_Float16)b1};
        wj0h[p]  = (v2h){(_Float16)Wc1[0 * HC + c], (_Float16)Wc1[0 * HC + c + 1]};
        wj1h[p]  = (v2h){(_Float16)Wc1[1 * HC + c], (_Float16)Wc1[1 * HC + c + 1]};
    }

    // B fragment: column n of Wc2. Halves 0..7 hold K=c0..c0+7 (8*half);
    // halves 8..15 (K=16..31) are zero padding (Hc = 16).
    v16h bf;
    #pragma unroll
    for (int t = 0; t < 16; ++t) {
        float w = (t < 8) ? Wc2[(c0 + t) * HC + n] : 0.f;
        bf[t] = (_Float16)w;
    }

    // C accumulator seed: C[m][n] = bc2[n]  (lane-constant splat) -> the
    // matrix core adds the bias for free.
    const float bc2n = bc2[n];
    v8f cbias;
    #pragma unroll
    for (int r = 0; r < 8; ++r) cbias[r] = bc2n;

    const v2h slope = (v2h){(_Float16)0.01f, (_Float16)0.01f};

    float acc = 0.f;    // sum_j sigmoid(A) * h2   (Wc3[n] applied after loop)

    // ---- j-tile loop: 32 tiles of 16 rows; unroll x2 to pipeline WMMAs --
    #pragma unroll 2
    for (int j0 = 0; j0 < N_; j0 += 16) {
        const int   j   = j0 + n;                       // A-fragment row M
        const float xj0 = x[(b * N_ + j) * 2 + 0];
        const float xj1 = x[(b * N_ + j) * 2 + 1];

        const _Float16 xj0h = (_Float16)xj0;
        const _Float16 xj1h = (_Float16)xj1;
        const v2h X0 = (v2h){xj0h, xj0h};
        const v2h X1 = (v2h){xj1h, xj1h};

        // A fragment: halves 0..7 = leaky(pj + pi + bc1) for channels
        // c0..c0+7 (K chunk by lane half); halves 8..15 = 0 (K padding).
        // Built entirely in packed f16: v_pk_fma_f16 + v_pk_max_f16.
        v16h af;
        #pragma unroll
        for (int p = 0; p < 4; ++p) {
            v2h v = X0 * wj0h[p] + (X1 * wj1h[p] + baseh[p]);
            v2h l = __builtin_elementwise_max(v, v * slope);    // leaky
            af[2 * p]     = l[0];
            af[2 * p + 1] = l[1];
            af[8 + 2 * p]     = (_Float16)0.f;
            af[8 + 2 * p + 1] = (_Float16)0.f;
        }

        v8f c = __builtin_amdgcn_wmma_f32_16x16x32_f16(
                    /*neg_a=*/false, af, /*neg_b=*/false, bf,
                    /*c_mod=*/(short)0, cbias, /*reuse_a=*/false, /*reuse_b=*/false);

        // Epilogue: D VGPR r holds row m = r + 8*half, column n = lane&15.
        // h2 = leaky(D) (bias already in C); 3 VALU per element.
        #pragma unroll
        for (int r = 0; r < 8; ++r) {
            float sg = srow[j0 + r + 8 * half];          // LDS broadcast read
            float h2 = fmaxf(c[r], 0.01f * c[r]);
            acc = fmaf(sg, h2, acc);
        }
    }

    acc *= Wc3[n];      // lane-constant weight, applied once

    // ---- wave reduction -------------------------------------------------
    #pragma unroll
    for (int off = 16; off > 0; off >>= 1) {
        acc  += __shfl_xor(acc,  off, 32);
        accA += __shfl_xor(accA, off, 32);
    }
    if (lane == 0) {
        out[(b * N_ + i) * 2 + 1] = node_ws[b * N_ + i] + acc + accA * bc3[0];
    }
}

// ---------------------------------------------------------------------------
extern "C" void kernel_launch(void* const* d_in, const int* in_sizes, int n_in,
                              void* d_out, int out_size, void* d_ws, size_t ws_size,
                              hipStream_t stream)
{
    const float* x   = (const float*)d_in[0];
    const float* Wn1 = (const float*)d_in[1];
    const float* bn1 = (const float*)d_in[2];
    const float* Wn2 = (const float*)d_in[3];
    const float* bn2 = (const float*)d_in[4];
    const float* Wn3 = (const float*)d_in[5];
    const float* bn3 = (const float*)d_in[6];
    const float* Wc1 = (const float*)d_in[7];
    const float* bc1 = (const float*)d_in[8];
    const float* Wc2 = (const float*)d_in[9];
    const float* bc2 = (const float*)d_in[10];
    const float* Wc3 = (const float*)d_in[11];
    const float* bc3 = (const float*)d_in[12];
    const float* Ap  = (const float*)d_in[13];

    float* out     = (float*)d_out;
    float* node_ws = (float*)d_ws;   // B*N floats = 32 KB

    // Kernel 1: one thread per (b,i) — 8192 threads.
    node_mlp_kernel<<<(B_ * N_) / 256, 256, 0, stream>>>(
        x, Wn1, bn1, Wn2, bn2, Wn3, bn3, node_ws, out);

    // Kernel 2: one wave32 per (b,i) — 8192 waves, 8 waves/block.
    coupling_kernel<<<(B_ * N_ * 32) / 256, 256, 0, stream>>>(
        x, Wc1, bc1, Wc2, bc2, Wc3, bc3, Ap, node_ws, out);
}
